// EMGEEGFusionEncoderv3Control_45217415692429
// MI455X (gfx1250) — matrix operand order repaired and checked by reading, not compiled, not run on hardware
//
#include <hip/hip_runtime.h>
#include <hip/hip_bf16.h>

typedef __attribute__((ext_vector_type(16))) _Float16 v16h;
typedef __attribute__((ext_vector_type(8)))  float    v8f;
typedef __attribute__((ext_vector_type(8)))  float    v8fv;

// ---------------------------------------------------------------------------
// WMMA GEMM: C[M x Nc] = relu?( A[M x KA] * B[KA x Nc] + bias )
// Requirements: M % 16 == 0, KA % 32 == 0, Nc % 64 == 0.
// A: fp32 row-major.  B: pre-packed fp16 fragments (see pack_weights_kernel).
// One wave computes a 16x64 tile (4 x 16x16 WMMA tiles, A fragment reused 4x).
// ---------------------------------------------------------------------------
__global__ void wmma_gemm4_kernel(const float* __restrict__ A,
                                  const _Float16* __restrict__ Bp,
                                  const float* __restrict__ bias,
                                  float* __restrict__ C,
                                  int M, int KA, int Nc, int do_relu)
{
    const int gtid    = blockIdx.x * blockDim.x + threadIdx.x;
    const int wave    = gtid >> 5;
    const int lane    = gtid & 31;
    const int tilesN4 = Nc >> 6;                 // 64-wide tiles
    const int tileM   = wave / tilesN4;
    const int tileN4  = wave - tileM * tilesN4;
    if (tileM * 16 >= M) return;                 // wave-uniform exit

    const int m0   = tileM << 4;
    const int n0   = tileN4 << 6;
    const int half = lane >> 4;                  // 0: lanes 0-15, 1: lanes 16-31
    const int lr   = lane & 15;                  // A row / B,C column within tile

    const int KB = KA >> 5;                      // k-blocks of 32
    const float* __restrict__ ap = A + (size_t)(m0 + lr) * KA;
    const v16h*  __restrict__ bp = (const v16h*)Bp;

    v8f acc0 = {}, acc1 = {}, acc2 = {}, acc3 = {};
    const int tnBase = tileN4 << 2;              // first 16-wide n-tile index

    for (int kb = 0; kb < KB; ++kb) {
        // A fragment: 16 contiguous floats per lane (k = 16*half + e, matches packed B).
        const int kbase = (kb << 5) + (half << 4);
        v8fv x0 = *(const v8fv*)(ap + kbase);
        v8fv x1 = *(const v8fv*)(ap + kbase + 8);
        v16h a;
        #pragma unroll
        for (int e = 0; e < 8; ++e) a[e] = (_Float16)x0[e];
        #pragma unroll
        for (int e = 0; e < 8; ++e) a[8 + e] = (_Float16)x1[e];

        // B fragments: one contiguous 32-byte v16h load per lane per tile.
        const v16h b0 = bp[(((tnBase + 0) * KB + kb) << 5) + lane];
        const v16h b1 = bp[(((tnBase + 1) * KB + kb) << 5) + lane];
        const v16h b2 = bp[(((tnBase + 2) * KB + kb) << 5) + lane];
        const v16h b3 = bp[(((tnBase + 3) * KB + kb) << 5) + lane];

        acc0 = __builtin_amdgcn_wmma_f32_16x16x32_f16(false, a, false, b0, (short)0, acc0, false, false);
        acc1 = __builtin_amdgcn_wmma_f32_16x16x32_f16(false, a, false, b1, (short)0, acc1, false, false);
        acc2 = __builtin_amdgcn_wmma_f32_16x16x32_f16(false, a, false, b2, (short)0, acc2, false, false);
        acc3 = __builtin_amdgcn_wmma_f32_16x16x32_f16(false, a, false, b3, (short)0, acc3, false, false);
    }

    // C/D layout: lane -> column (tile col lr), VGPR r -> row m0 + r + half*8.
    #pragma unroll
    for (int t = 0; t < 4; ++t) {
        const v8f av = (t == 0) ? acc0 : (t == 1) ? acc1 : (t == 2) ? acc2 : acc3;
        const int col = n0 + (t << 4) + lr;
        const float bv = bias ? bias[col] : 0.0f;
        #pragma unroll
        for (int r = 0; r < 8; ++r) {
            const int m = m0 + r + (half << 3);
            float v = av[r] + bv;
            if (do_relu) v = fmaxf(v, 0.0f);
            C[(size_t)m * Nc + col] = v;
        }
    }
}

// ---------------------------------------------------------------------------
// Pack fp32 weights [Kreal x Nc] into fp16 WMMA B-fragment order, zero-padding
// K up to KB*32.  Layout: P[((tn*KB + kb)*32 + lane)*16 + e],
// where value = B[(kb*32 + half*16 + e) * Nc + tn*16 + lr].
// ---------------------------------------------------------------------------
__global__ void pack_weights_kernel(const float* __restrict__ B,
                                    _Float16* __restrict__ P,
                                    int Kreal, int KB, int Nc)
{
    const int total  = (Nc >> 4) * KB * 512;
    const int stride = gridDim.x * blockDim.x;
    for (int idx = blockIdx.x * blockDim.x + threadIdx.x; idx < total; idx += stride) {
        const int e    = idx & 15;
        const int lane = (idx >> 4) & 31;
        const int rest = idx >> 9;
        const int kb   = rest % KB;
        const int tn   = rest / KB;
        const int half = lane >> 4;
        const int lr   = lane & 15;
        const int k    = (kb << 5) + (half << 4) + e;
        const int n    = (tn << 4) + lr;
        P[idx] = (k < Kreal) ? (_Float16)B[(size_t)k * Nc + n] : (_Float16)0.0f;
    }
}

// xpad[i][0..31] = (f < 10) ? x[i][f] + agg[i][f] : 0
__global__ void pack_pad_kernel(const float* __restrict__ x,
                                const float* __restrict__ agg,
                                float* __restrict__ xpad, int N)
{
    const size_t total  = (size_t)N * 32;
    const size_t stride = (size_t)gridDim.x * blockDim.x;
    for (size_t idx = (size_t)blockIdx.x * blockDim.x + threadIdx.x; idx < total; idx += stride) {
        const size_t i = idx >> 5;
        const int    f = (int)(idx & 31);
        xpad[idx] = (f < 10) ? (x[i * 10 + f] + agg[i * 10 + f]) : 0.0f;
    }
}

__global__ void add_kernel(const float* __restrict__ a, const float* __restrict__ b,
                           float* __restrict__ o, size_t n)
{
    const size_t stride = (size_t)gridDim.x * blockDim.x;
    for (size_t i = (size_t)blockIdx.x * blockDim.x + threadIdx.x; i < n; i += stride)
        o[i] = a[i] + b[i];
}

__global__ void fill_kernel(float* __restrict__ p, float val, size_t n)
{
    const size_t stride = (size_t)gridDim.x * blockDim.x;
    for (size_t i = (size_t)blockIdx.x * blockDim.x + threadIdx.x; i < n; i += stride)
        p[i] = val;
}

// agg[dst[e]][:] += x[src[e]][:]   (wave per edge, lanes across features)
__global__ void scatter_add_kernel(const float* __restrict__ x,
                                   const int* __restrict__ src,
                                   const int* __restrict__ dst,
                                   float* __restrict__ agg,
                                   int E, int F)
{
    const int gtid  = blockIdx.x * blockDim.x + threadIdx.x;
    const int lane  = gtid & 31;
    const int nwave = (gridDim.x * blockDim.x) >> 5;
    for (int e = gtid >> 5; e < E; e += nwave) {
        const int s = src[e], d = dst[e];
        const float* xs = x   + (size_t)s * F;
        float*       ad = agg + (size_t)d * F;
        for (int f = lane; f < F; f += 32)
            atomicAdd(&ad[f], xs[f]);
    }
}

// alpha_s[i] = dot(h[i], a_src); alpha_d[i] = dot(h[i], a_dst)
__global__ void alpha_kernel(const float* __restrict__ h,
                             const float* __restrict__ asrc,
                             const float* __restrict__ adst,
                             float* __restrict__ als, float* __restrict__ ald,
                             int N, int D)
{
    const int gtid  = blockIdx.x * blockDim.x + threadIdx.x;
    const int lane  = gtid & 31;
    const int nwave = (gridDim.x * blockDim.x) >> 5;
    for (int i = gtid >> 5; i < N; i += nwave) {
        const float* hp = h + (size_t)i * D;
        float ss = 0.0f, sd = 0.0f;
        for (int f = lane; f < D; f += 32) {
            const float v = hp[f];
            ss += v * asrc[f];
            sd += v * adst[f];
        }
        #pragma unroll
        for (int off = 16; off > 0; off >>= 1) {
            ss += __shfl_down(ss, off);
            sd += __shfl_down(sd, off);
        }
        if (lane == 0) { als[i] = ss; ald[i] = sd; }
    }
}

__device__ inline float leaky02(float x) { return x > 0.0f ? x : 0.2f * x; }

__device__ inline void atomicMaxFloat(float* addr, float value)
{
    if (value >= 0.0f)
        atomicMax((int*)addr, __float_as_int(value));
    else
        atomicMin((unsigned int*)addr, __float_as_uint(value));
}

// segment max of leaky_relu(as[s] + ad[d]) over dst  (E edges + N self loops)
__global__ void edge_max_kernel(const float* __restrict__ als,
                                const float* __restrict__ ald,
                                const int* __restrict__ src,
                                const int* __restrict__ dst,
                                float* __restrict__ mx, int E, int N)
{
    const size_t total  = (size_t)E + N;
    const size_t stride = (size_t)gridDim.x * blockDim.x;
    for (size_t idx = (size_t)blockIdx.x * blockDim.x + threadIdx.x; idx < total; idx += stride) {
        int s, d;
        if (idx < (size_t)E) { s = src[idx]; d = dst[idx]; }
        else                 { s = d = (int)(idx - E); }
        atomicMaxFloat(&mx[d], leaky02(als[s] + ald[d]));
    }
}

// e = exp(logit - mx[d]); ssum[d] += e; acc[d][:] += e * h[s][:]
__global__ void edge_acc_kernel(const float* __restrict__ als,
                                const float* __restrict__ ald,
                                const float* __restrict__ h,
                                const int* __restrict__ src,
                                const int* __restrict__ dst,
                                const float* __restrict__ mx,
                                float* __restrict__ ssum,
                                float* __restrict__ acc,
                                int E, int N, int F)
{
    const int gtid  = blockIdx.x * blockDim.x + threadIdx.x;
    const int lane  = gtid & 31;
    const int nwave = (gridDim.x * blockDim.x) >> 5;
    const int total = E + N;
    for (int idx = gtid >> 5; idx < total; idx += nwave) {
        int s, d;
        if (idx < E) { s = src[idx]; d = dst[idx]; }
        else         { s = d = idx - E; }
        const float ev = expf(leaky02(als[s] + ald[d]) - mx[d]);
        if (lane == 0) atomicAdd(&ssum[d], ev);
        const float* hs = h   + (size_t)s * F;
        float*       ac = acc + (size_t)d * F;
        for (int f = lane; f < F; f += 32)
            atomicAdd(&ac[f], ev * hs[f]);
    }
}

// out = relu?( acc / (ssum + 1e-16) + bias )
__global__ void finalize_kernel(const float* __restrict__ acc,
                                const float* __restrict__ ssum,
                                const float* __restrict__ bias,
                                float* __restrict__ out,
                                int N, int F, int do_relu)
{
    const size_t total  = (size_t)N * F;
    const size_t stride = (size_t)gridDim.x * blockDim.x;
    for (size_t idx = (size_t)blockIdx.x * blockDim.x + threadIdx.x; idx < total; idx += stride) {
        const int i = (int)(idx / F);
        const int f = (int)(idx - (size_t)i * F);
        float v = acc[idx] / (ssum[i] + 1e-16f) + bias[f];
        if (do_relu) v = fmaxf(v, 0.0f);
        out[idx] = v;
    }
}

// ---------------------------------------------------------------------------
// Host orchestration
// ---------------------------------------------------------------------------
static inline void launch_gemm(const float* A, const _Float16* Bp, const float* bias,
                               float* C, int M, int KA, int Nc, int relu, hipStream_t s)
{
    const int waves   = (M / 16) * (Nc / 64);
    const int threads = 256;
    const int blocks  = (waves * 32 + threads - 1) / threads;
    hipLaunchKernelGGL(wmma_gemm4_kernel, dim3(blocks), dim3(threads), 0, s,
                       A, Bp, bias, C, M, KA, Nc, relu);
}

static inline void launch_pack(const float* B, _Float16* P, int Kreal, int KB, int Nc,
                               hipStream_t s)
{
    const int total = (Nc / 16) * KB * 512;
    hipLaunchKernelGGL(pack_weights_kernel, dim3((total + 255) / 256), dim3(256), 0, s,
                       B, P, Kreal, KB, Nc);
}

static inline void launch_fill(float* p, float v, size_t n, hipStream_t s)
{
    hipLaunchKernelGGL(fill_kernel, dim3(2048), dim3(256), 0, s, p, v, n);
}

extern "C" void kernel_launch(void* const* d_in, const int* in_sizes, int n_in,
                              void* d_out, int out_size, void* d_ws, size_t ws_size,
                              hipStream_t stream)
{
    const float* x        = (const float*)d_in[0];
    const int*   ei       = (const int*)  d_in[1];
    const float* gin1_w1  = (const float*)d_in[3];
    const float* gin1_b1  = (const float*)d_in[4];
    const float* gin1_w2  = (const float*)d_in[5];
    const float* gin1_b2  = (const float*)d_in[6];
    const float* gin2_w1  = (const float*)d_in[7];
    const float* gin2_b1  = (const float*)d_in[8];
    const float* gin2_w2  = (const float*)d_in[9];
    const float* gin2_b2  = (const float*)d_in[10];
    const float* gat1_w   = (const float*)d_in[11];
    const float* gat1_as  = (const float*)d_in[12];
    const float* gat1_ad  = (const float*)d_in[13];
    const float* gat1_b   = (const float*)d_in[14];
    const float* gat2_w   = (const float*)d_in[15];
    const float* gat2_as  = (const float*)d_in[16];
    const float* gat2_ad  = (const float*)d_in[17];
    const float* gat2_b   = (const float*)d_in[18];

    const int N = in_sizes[0] / 10;
    const int E = in_sizes[1] / 2;
    const int* src = ei;
    const int* dst = ei + E;

    float* ws   = (float*)d_ws;
    float* bufA = ws;                         // N x 256
    float* bufB = bufA + (size_t)N * 256;     // N x 256
    float* bufC = bufB + (size_t)N * 256;     // N x 256 (agg / accumulator)
    float* xpad = bufC + (size_t)N * 256;     // N x 32 (padded GIN1 input)
    float* als  = xpad + (size_t)N * 32;      // N
    float* ald  = als + N;                    // N
    float* mx   = ald + N;                    // N
    float* ssum = mx  + N;                    // N

    uintptr_t pw = ((uintptr_t)(ssum + N) + 63) & ~(uintptr_t)63;
    _Float16* p_gin1w1 = (_Float16*)pw;            // 32  x 64
    _Float16* p_gin1w2 = p_gin1w1 + 32 * 64;       // 64  x 64
    _Float16* p_gin2w1 = p_gin1w2 + 64 * 64;       // 64  x 128
    _Float16* p_gin2w2 = p_gin2w1 + 64 * 128;      // 128 x 128
    _Float16* p_gat1w  = p_gin2w2 + 128 * 128;     // 128 x 256
    _Float16* p_gat2w  = p_gat1w  + 128 * 256;     // 256 x 128

    // Pre-pack all weights into WMMA fragment order (tiny kernels).
    launch_pack(gin1_w1, p_gin1w1, 10,  1, 64,  stream);
    launch_pack(gin1_w2, p_gin1w2, 64,  2, 64,  stream);
    launch_pack(gin2_w1, p_gin2w1, 64,  2, 128, stream);
    launch_pack(gin2_w2, p_gin2w2, 128, 4, 128, stream);
    launch_pack(gat1_w,  p_gat1w,  128, 4, 256, stream);
    launch_pack(gat2_w,  p_gat2w,  256, 8, 128, stream);

    const int ethreads = 256;
    const int eblocks  = 4096;
    float* out = (float*)d_out;

    // ---- GIN layer 1 (10 -> 64 -> 64) ----
    launch_fill(bufC, 0.0f, (size_t)N * 10, stream);
    hipLaunchKernelGGL(scatter_add_kernel, dim3(eblocks), dim3(ethreads), 0, stream,
                       x, src, dst, bufC, E, 10);
    hipLaunchKernelGGL(pack_pad_kernel, dim3(2048), dim3(256), 0, stream,
                       x, bufC, xpad, N);
    launch_gemm(xpad, p_gin1w1, gin1_b1, bufA, N, 32, 64, 1, stream);       // relu
    launch_gemm(bufA, p_gin1w2, gin1_b2, bufB, N, 64, 64, 1, stream);       // + outer relu

    // ---- GIN layer 2 (64 -> 128 -> 128) ----
    launch_fill(bufC, 0.0f, (size_t)N * 64, stream);
    hipLaunchKernelGGL(scatter_add_kernel, dim3(eblocks), dim3(ethreads), 0, stream,
                       bufB, src, dst, bufC, E, 64);
    hipLaunchKernelGGL(add_kernel, dim3(2048), dim3(256), 0, stream,
                       bufB, bufC, bufC, (size_t)N * 64);
    launch_gemm(bufC, p_gin2w1, gin2_b1, bufA, N, 64, 128, 1, stream);      // relu
    launch_gemm(bufA, p_gin2w2, gin2_b2, bufB, N, 128, 128, 0, stream);     // h2

    // ---- GAT layer 1 (128 -> 256, relu after) ----
    launch_gemm(bufB, p_gat1w, nullptr, bufA, N, 128, 256, 0, stream);      // hw
    hipLaunchKernelGGL(alpha_kernel, dim3(2048), dim3(256), 0, stream,
                       bufA, gat1_as, gat1_ad, als, ald, N, 256);
    launch_fill(mx, -__builtin_huge_valf(), (size_t)N, stream);
    hipLaunchKernelGGL(edge_max_kernel, dim3(4096), dim3(256), 0, stream,
                       als, ald, src, dst, mx, E, N);
    launch_fill(ssum, 0.0f, (size_t)N, stream);
    launch_fill(bufC, 0.0f, (size_t)N * 256, stream);
    hipLaunchKernelGGL(edge_acc_kernel, dim3(eblocks), dim3(ethreads), 0, stream,
                       als, ald, bufA, src, dst, mx, ssum, bufC, E, N, 256);
    hipLaunchKernelGGL(finalize_kernel, dim3(4096), dim3(256), 0, stream,
                       bufC, ssum, gat1_b, bufB, N, 256, 1);                // h3 (relu)

    // ---- GAT layer 2 (256 -> 128) ----
    launch_gemm(bufB, p_gat2w, nullptr, bufA, N, 256, 128, 0, stream);      // hw2
    hipLaunchKernelGGL(alpha_kernel, dim3(2048), dim3(256), 0, stream,
                       bufA, gat2_as, gat2_ad, als, ald, N, 128);
    launch_fill(mx, -__builtin_huge_valf(), (size_t)N, stream);
    hipLaunchKernelGGL(edge_max_kernel, dim3(4096), dim3(256), 0, stream,
                       als, ald, src, dst, mx, E, N);
    launch_fill(ssum, 0.0f, (size_t)N, stream);
    launch_fill(bufC, 0.0f, (size_t)N * 128, stream);
    hipLaunchKernelGGL(edge_acc_kernel, dim3(eblocks), dim3(ethreads), 0, stream,
                       als, ald, bufA, src, dst, mx, ssum, bufC, E, N, 128);
    hipLaunchKernelGGL(finalize_kernel, dim3(4096), dim3(256), 0, stream,
                       bufC, ssum, gat2_b, out, N, 128, 0);
}
